// MultiheadSelfAttention_81466939671053
// MI455X (gfx1250) — matrix operand order repaired and checked by reading, not compiled, not run on hardware
//
#include <hip/hip_runtime.h>
#include <hip/hip_bf16.h>

// ---------------------------------------------------------------------------
// MI455X (gfx1250) multihead self-attention.
//  * all matmuls: v_wmma_f32_16x16x32_bf16 (wave32 WMMA, f32 accum)
//  * GEMM B-panels staged into LDS by the Tensor Data Mover
//    (tensor_load_to_lds + s_wait_tensorcnt), double buffered
//  * fused causal flash-attention with online softmax
// ---------------------------------------------------------------------------

typedef __attribute__((ext_vector_type(16))) __bf16     v16bf;
typedef __attribute__((ext_vector_type(8)))  float      v8f;
typedef __attribute__((ext_vector_type(4)))  unsigned   u32x4;
typedef __attribute__((ext_vector_type(4)))  unsigned   ui32x4;
typedef __attribute__((ext_vector_type(8)))  int        i32x8;
typedef __attribute__((ext_vector_type(4)))  int        i32x4;

#define BATCH  4
#define HEADS  16
#define SEQ    2048
#define DK     64
#define DM     1024
#define MTOT   (BATCH * SEQ)   // 8192

union AFrag { v16bf v; u32x4 q[2]; };
union BFrag { v16bf v; u32x4 q[2]; };

__device__ __forceinline__ __bf16 f2bf(float f) {
    unsigned u = __builtin_bit_cast(unsigned, f);
    unsigned r = (u + 0x7FFFu + ((u >> 16) & 1u)) >> 16;   // round-nearest-even
    return __builtin_bit_cast(__bf16, (unsigned short)r);
}

__device__ __forceinline__ v8f wmma_bf16(v16bf a, v16bf b, v8f c) {
    // (neg_a, A, neg_b, B, c_mod, C, reuse_a, reuse_b)
    return __builtin_amdgcn_wmma_f32_16x16x32_bf16(false, a, false, b,
                                                   (short)0, c, false, false);
}

// A-fragment: 16x32 bf16 tile at `base` (row-major, stride ld).
// ISA 7.12.2: lanes 0-15 row=lane, K={0..7,16..23}; lanes 16-31 K={8..15,24..31}.
__device__ __forceinline__ AFrag load_a(const __bf16* base, int ld, int lane) {
    int row  = lane & 15;
    int koff = (lane < 16) ? 0 : 8;
    const __bf16* p = base + (size_t)row * ld + koff;
    AFrag a;
    a.q[0] = *(const u32x4*)(p);        // K = koff .. koff+7
    a.q[1] = *(const u32x4*)(p + 16);   // K = koff+16 .. koff+23
    return a;
}

// B-fragment: 32x16 bf16 tile (K x N, row-major, stride ld). Lane l = col N=l%16, K={0..7,16..23} / {8..15,24..31}.
__device__ __forceinline__ BFrag load_b(const __bf16* base, int ld, int lane) {
    const __bf16* p = base + (lane & 15);
    const int h8 = (lane < 16) ? 0 : 8;
    BFrag b;
#pragma unroll
    for (int s = 0; s < 8; ++s) {
        b.v[s]     = p[(size_t)(h8 + s) * ld];
        b.v[s + 8] = p[(size_t)(16 + h8 + s) * ld];
    }
    return b;
}

// ---------------------------------------------------------------------------
// TDM: DMA one 32-row x 64-col bf16 panel (4 KB) from global into LDS.
// D# per CDNA5 ISA. data_size=4B (bf16 pairs):
//   row = 32 dwords, row stride = DM*2B = 512 dwords, 32 rows.
// Caller must guard to a single wave and later s_wait_tensorcnt + barrier.
// ---------------------------------------------------------------------------
__device__ __forceinline__ void tdm_stage_panel(const __bf16* gsrc, __bf16* ldst) {
    unsigned lds_byte = (unsigned)(size_t)ldst;          // LDS aperture: addr[31:0]
    unsigned long long ga = (unsigned long long)(size_t)gsrc;
    ui32x4 g0 = {};
    g0[0] = 1u;                                          // count=1 (valid, user)
    g0[1] = lds_byte;                                    // lds_addr
    g0[2] = (unsigned)ga;                                // global_addr[31:0]
    g0[3] = (unsigned)((ga >> 32) & 0x01FFFFFFu)         // global_addr[56:32]
          | (2u << 30);                                  // type=2 ("image")
    i32x8 g1 = {};
    g1[0] = (int)(2u << 16);      // data_size = 2 -> 4 bytes
    g1[1] = (int)(512u << 16);    // tensor_dim0 lo16 = 512 dwords (full row)
    g1[2] = (int)(1024u << 16);   // tensor_dim0 hi=0 | tensor_dim1 lo16 = 1024 rows
    g1[3] = (int)(32u << 16);     // tensor_dim1 hi=0 | tile_dim0 = 32 dwords
    g1[4] = 32;                   // tile_dim1 = 32 rows, tile_dim2 = 0
    g1[5] = 512;                  // tensor_dim0_stride = 512 dwords
    g1[6] = 0;                    // stride hi / tensor_dim1_stride lo
    g1[7] = 0;
    i32x4 gz4 = {};
    i32x8 gz8 = {};
    // amdgpu-toolchain (clang-23) 6-arg form; cpol = 0
    __builtin_amdgcn_tensor_load_to_lds(g0, g1, gz4, gz4, gz8, 0);
}

// ---------------------------------------------------------------------------
// Stage 1/2: conversions
// ---------------------------------------------------------------------------
__global__ void cvt_f32_bf16(const float* __restrict__ in,
                             __bf16* __restrict__ out, int n) {
    int i = blockIdx.x * blockDim.x + threadIdx.x;
    if (i < n) out[i] = f2bf(in[i]);
}

// W [out,in] f32 -> Wt [in,out] bf16
__global__ void w_transpose_bf16(const float* __restrict__ W,
                                 __bf16* __restrict__ Wt) {
    int i = blockIdx.x * blockDim.x + threadIdx.x;  // i = n*DM + k
    int n = i >> 10, k = i & (DM - 1);
    Wt[(size_t)k * DM + n] = f2bf(W[i]);
}

// ---------------------------------------------------------------------------
// Stage 3/5: GEMM  Y[8192 x 1024] = A[8192 x 1024] * B[1024 x 1024]
// Block = 128 threads (4 waves); block tile 128x64 (wave -> 32x64).
// B panel (32x64 per K-step) staged into LDS by TDM, double buffered;
// all 4 waves read B fragments as strided 16-bit LDS loads.
// MODE: 0 = bf16 per-head [b,h,s,d] (Q/V); 1 = per-head transposed [b,h,d,s]
// (K); 2 = f32 row-major (final output).
// ---------------------------------------------------------------------------
template <int MODE>
__global__ void __launch_bounds__(128)
gemm_wmma(const __bf16* __restrict__ A, const __bf16* __restrict__ Bm,
          void* __restrict__ outp) {
    __shared__ __bf16 Bl[2][32 * 64];                  // 2 x 4 KB panels

    const int lane  = threadIdx.x & 31;
    const int wave  = threadIdx.x >> 5;
    const int mBase = blockIdx.y * 128 + wave * 32;
    const int nBase = blockIdx.x * 64;
    const int half8 = (lane < 16) ? 0 : 8;
    const int col   = lane & 15;

    v8f acc[2][4] = {};

    // prime buffer 0
    if (threadIdx.x == 0) tdm_stage_panel(Bm + nBase, &Bl[0][0]);
    if (threadIdx.x < 32) __builtin_amdgcn_s_wait_tensorcnt(0);
    __syncthreads();

    for (int k = 0; k < DM; k += 32) {
        const int buf = (k >> 5) & 1;
        // kick off next panel DMA while we compute this one
        if (k + 32 < DM && threadIdx.x == 0)
            tdm_stage_panel(Bm + (size_t)(k + 32) * DM + nBase, &Bl[buf ^ 1][0]);

        AFrag a0 = load_a(A + (size_t)mBase * DM + k, DM, lane);
        AFrag a1 = load_a(A + (size_t)(mBase + 16) * DM + k, DM, lane);
#pragma unroll
        for (int t = 0; t < 4; ++t) {
            BFrag b = load_b(&Bl[buf][t * 16], 64, lane);   // 16 strided 2-byte LDS loads
            acc[0][t] = wmma_bf16(a0.v, b.v, acc[0][t]);
            acc[1][t] = wmma_bf16(a1.v, b.v, acc[1][t]);
        }

        // issuing wave waits for the DMA; barrier releases the panel to all
        // waves and protects the buffer we overwrite next iteration (WAR).
        if (threadIdx.x < 32) __builtin_amdgcn_s_wait_tensorcnt(0);
        __syncthreads();
    }

#pragma unroll
    for (int mt = 0; mt < 2; ++mt)
#pragma unroll
        for (int t = 0; t < 4; ++t)
#pragma unroll
            for (int r = 0; r < 8; ++r) {
                int rg = mBase + mt * 16 + r + half8;   // global row (token)
                int cg = nBase + t * 16 + col;          // global col (feature)
                float val = acc[mt][t][r];
                if (MODE == 2) {
                    ((float*)outp)[(size_t)rg * DM + cg] = val;
                } else {
                    int b = rg >> 11, s = rg & (SEQ - 1);
                    int h = cg >> 6,  d = cg & (DK - 1);
                    __bf16* o = (__bf16*)outp;
                    if (MODE == 1)   // K transposed per head: [b,h,d,s]
                        o[(((size_t)b * HEADS + h) * DK + d) * SEQ + s] = f2bf(val);
                    else             // Q / V per head: [b,h,s,d]
                        o[(((size_t)b * HEADS + h) * SEQ + s) * DK + d] = f2bf(val);
                }
            }
}

// ---------------------------------------------------------------------------
// Stage 4: flash attention. One wave per (b, h, 16-query tile).
// Per 32-key step: S(16x32) via 4 WMMAs, online softmax, P through LDS
// (C-layout -> A-fragment layout), O += P*V via 4 WMMAs.
// ---------------------------------------------------------------------------
__global__ void __launch_bounds__(32)
flash_attn(const __bf16* __restrict__ Qh, const __bf16* __restrict__ Kt,
           const __bf16* __restrict__ Vh, __bf16* __restrict__ Ob) {
    const int lane  = threadIdx.x & 31;
    const int qt    = blockIdx.x;
    const int h     = blockIdx.y;
    const int b     = blockIdx.z;
    const int qbase = qt * 16;
    const int half8 = (lane < 16) ? 0 : 8;
    const int col   = lane & 15;
    const int koff  = (lane < 16) ? 0 : 8;
    const int arow  = lane & 15;

    const size_t bh = (size_t)b * HEADS + h;
    const __bf16* Qp = Qh + (bh * SEQ + qbase) * DK;
    const __bf16* Kp = Kt + bh * DK * SEQ;          // [d][s]
    const __bf16* Vp = Vh + bh * SEQ * DK;          // [s][d]

    AFrag aQ0 = load_a(Qp, DK, lane);               // K-dim 0..31
    AFrag aQ1 = load_a(Qp + 32, DK, lane);          // K-dim 32..63

    v8f acc[4] = {};
    float m[8], l[8];
#pragma unroll
    for (int r = 0; r < 8; ++r) { m[r] = -1e30f; l[r] = 0.f; }

    __shared__ __bf16 Pl[16 * 32];

    for (int kv = 0; kv < qbase + 16; kv += 32) {
        // ---- S = Q K^T (two 16x16 tiles: cols kv..kv+15 and kv+16..kv+31)
        v8f s0 = {}, s1 = {};
        const __bf16* kp0 = Kp + kv;                             // d = 0..31
        const __bf16* kp1 = Kp + (size_t)32 * SEQ + kv;          // d = 32..63
        BFrag bk;
        bk = load_b(kp0, SEQ, lane);
        s0 = wmma_bf16(aQ0.v, bk.v, s0);
        bk = load_b(kp1, SEQ, lane);
        s0 = wmma_bf16(aQ1.v, bk.v, s0);
        bk = load_b(kp0 + 16, SEQ, lane);
        s1 = wmma_bf16(aQ0.v, bk.v, s1);
        bk = load_b(kp1 + 16, SEQ, lane);
        s1 = wmma_bf16(aQ1.v, bk.v, s1);

        // ---- scale, causal mask, online softmax (rows live in 16-lane halves)
        float p0[8], p1[8];
#pragma unroll
        for (int r = 0; r < 8; ++r) {
            int qrow = qbase + r + half8;
            float x0 = s0[r] * 0.125f;                 // 1/sqrt(64)
            float x1 = s1[r] * 0.125f;
            if (kv + col > qrow)      x0 = -1e30f;
            if (kv + 16 + col > qrow) x1 = -1e30f;
            float t = fmaxf(x0, x1);
#pragma unroll
            for (int off = 1; off < 16; off <<= 1)
                t = fmaxf(t, __shfl_xor(t, off, 32));
            float mn = fmaxf(m[r], t);
            float e0 = __expf(x0 - mn), e1 = __expf(x1 - mn);
            float rs = e0 + e1;
#pragma unroll
            for (int off = 1; off < 16; off <<= 1)
                rs += __shfl_xor(rs, off, 32);
            float al = __expf(m[r] - mn);
            l[r] = l[r] * al + rs;
            m[r] = mn;
            p0[r] = e0; p1[r] = e1;
#pragma unroll
            for (int t4 = 0; t4 < 4; ++t4) acc[t4][r] *= al;
        }

        // ---- P (C-layout) -> LDS -> A-fragment layout
#pragma unroll
        for (int r = 0; r < 8; ++r) {
            int row = r + half8;
            Pl[row * 32 + col]      = f2bf(p0[r]);
            Pl[row * 32 + col + 16] = f2bf(p1[r]);
        }
        asm volatile("s_wait_dscnt 0" ::: "memory");   // cross-lane LDS visibility
        AFrag aP;
        const __bf16* pp = &Pl[arow * 32 + koff];
        aP.q[0] = *(const u32x4*)pp;
        aP.q[1] = *(const u32x4*)(pp + 16);

        // ---- O += P * V   (V rows kv..kv+31, 4 N-tiles of 16 dims)
#pragma unroll
        for (int t4 = 0; t4 < 4; ++t4) {
            BFrag bv = load_b(Vp + (size_t)kv * DK + t4 * 16, DK, lane);
            acc[t4] = wmma_bf16(aP.v, bv.v, acc[t4]);
        }
    }

    // ---- normalize, write O as bf16 row-major [b, s, h*64+d]
#pragma unroll
    for (int t4 = 0; t4 < 4; ++t4)
#pragma unroll
        for (int r = 0; r < 8; ++r) {
            float v = acc[t4][r] / l[r];
            size_t row = (size_t)b * SEQ + qbase + r + half8;
            Ob[row * DM + h * DK + t4 * 16 + col] = f2bf(v);
        }
}

// ---------------------------------------------------------------------------
extern "C" void kernel_launch(void* const* d_in, const int* in_sizes, int n_in,
                              void* d_out, int out_size, void* d_ws, size_t ws_size,
                              hipStream_t stream) {
    const float* x  = (const float*)d_in[0];
    const float* Wq = (const float*)d_in[1];
    const float* Wk = (const float*)d_in[2];
    const float* Wv = (const float*)d_in[3];
    const float* Wo = (const float*)d_in[4];
    float* out = (float*)d_out;

    char* ws = (char*)d_ws;
    const size_t XB = (size_t)MTOT * DM * 2;           // 16 MB
    const size_t WT = (size_t)DM * DM * 2;             // 2 MB
    __bf16* Xb  = (__bf16*)ws;              ws += XB;
    __bf16* Wqt = (__bf16*)ws;              ws += WT;
    __bf16* Wkt = (__bf16*)ws;              ws += WT;
    __bf16* Wvt = (__bf16*)ws;              ws += WT;
    __bf16* Wot = (__bf16*)ws;              ws += WT;
    __bf16* Qh  = (__bf16*)ws;              ws += XB;  // [b,h,s,d]
    __bf16* Ktr = (__bf16*)ws;              ws += XB;  // [b,h,d,s]
    __bf16* Vh  = (__bf16*)ws;              ws += XB;  // [b,h,s,d]
    __bf16* Ob  = (__bf16*)ws;              ws += XB;  // [b*s, 1024]

    // Stage 1: x -> bf16
    cvt_f32_bf16<<<(MTOT * DM) / 256, 256, 0, stream>>>(x, Xb, MTOT * DM);
    // Stage 2: weights -> transposed bf16
    w_transpose_bf16<<<(DM * DM) / 256, 256, 0, stream>>>(Wq, Wqt);
    w_transpose_bf16<<<(DM * DM) / 256, 256, 0, stream>>>(Wk, Wkt);
    w_transpose_bf16<<<(DM * DM) / 256, 256, 0, stream>>>(Wv, Wvt);
    w_transpose_bf16<<<(DM * DM) / 256, 256, 0, stream>>>(Wo, Wot);

    // Stage 3: Q/K/V projections (WMMA + TDM-staged B panels)
    dim3 gg(DM / 64, MTOT / 128);
    gemm_wmma<0><<<gg, 128, 0, stream>>>(Xb, Wqt, Qh);
    gemm_wmma<1><<<gg, 128, 0, stream>>>(Xb, Wkt, Ktr);
    gemm_wmma<0><<<gg, 128, 0, stream>>>(Xb, Wvt, Vh);

    // Stage 4: fused causal flash attention (WMMA + online softmax)
    flash_attn<<<dim3(SEQ / 16, HEADS, BATCH), 32, 0, stream>>>(Qh, Ktr, Vh, Ob);

    // Stage 5: output projection -> f32
    gemm_wmma<2><<<gg, 128, 0, stream>>>(Ob, Wot, out);
}
